// EditModel_1675037245647
// MI455X (gfx1250) — compile-verified, matrix operand order checked by reading, not run on hardware
//
#include <hip/hip_runtime.h>
#include <hip/hip_bf16.h>
#include <math.h>

// ---------------------------------------------------------------------------
// EditModel fused kernel for MI455X (gfx1250, wave32)
//
// logit[x,y,b,v] = (prior[x,b]+modern[y,b])·W[v] + bias[v]
//                = A[x,b,v] + C[y,b,v]          (separable!)
//
// Phase 1: 4 GEMMs (8192 x 53 x 256) via v_wmma_f32_16x16x32_f16
//          -> ws layout [mat][x][vpad=64][b=128] fp32, bias folded into C.
// Phase 2: per (x,y,b): online logsumexp over v for both heads, emit
//          del/end (v=52) and sub/ins (v=target) log-probs, masked.
// ---------------------------------------------------------------------------

typedef __attribute__((ext_vector_type(16))) _Float16 v16h;
typedef __attribute__((ext_vector_type(8)))  float    v8f;

#define LX 64
#define LY 64
#define BB 128
#define DD 256
#define VOUT 53
#define VPAD 64
#define MAT_STRIDE (LX * VPAD * BB)   // 524288 floats per matrix

// --------------------------- Phase 1: WMMA GEMM ----------------------------
// tile id decomposition: [mat(2b)][m_tile(9b)][col_tile(2b)] ; 8192 tiles,
// 1024 blocks x 8 waves. mat: 0=A_sub 1=A_ins 2=C_sub 3=C_ins.
__global__ __launch_bounds__(256) void edit_gemm_wmma(
    const float* __restrict__ prior, const float* __restrict__ modern,
    const float* __restrict__ Wsub,  const float* __restrict__ bsub,
    const float* __restrict__ Wins,  const float* __restrict__ bins,
    float* __restrict__ ws)
{
  const int lane = threadIdx.x & 31;
  const int wave = threadIdx.x >> 5;
  const int tile = blockIdx.x * 8 + wave;

  const int col_tile = tile & 3;
  const int m_tile   = (tile >> 2) & 511;
  const int mat      = tile >> 11;          // 0..3
  const int head     = mat & 1;             // 0=sub 1=ins
  const int is_mod   = mat >> 1;            // 0=prior(A) 1=modern(C)

  const float* __restrict__ ctx  = is_mod ? modern : prior;
  const float* __restrict__ W    = head ? Wins : Wsub;
  const float* __restrict__ bias = head ? bins : bsub;
  float* __restrict__ out        = ws + (size_t)mat * MAT_STRIDE;

  const int m_base = m_tile * 16;           // rows (x*128+b); tile never crosses x
  const int r16    = lane & 15;
  const int khalf  = lane >> 4;             // 0 or 1

  // A fragment source: row m = m_base + r16 of ctx (m*256 + k)
  const float* __restrict__ arow = ctx + (size_t)(m_base + r16) * DD;

  // B fragment source: column v = col_tile*16 + r16 of W (v*256 + k)
  const int   vcol   = col_tile * 16 + r16;
  const int   vsafe  = (vcol < VOUT) ? vcol : 0;
  const float bscale = (vcol < VOUT) ? 1.0f : 0.0f;   // zero padded columns
  const float* __restrict__ brow = W + (size_t)vsafe * DD;

  v8f acc = {};
  #pragma unroll
  for (int kk = 0; kk < DD; kk += 32) {
    v16h afrag, bfrag;
    #pragma unroll
    for (int p = 0; p < 8; ++p) {
      // f16 A/B layout: vgpr p holds K = {2p, 2p+1} (p<4) or {16+2(p-4), ...},
      // with lanes 16..31 shifted by +8 in K.
      const int k = kk + ((p < 4) ? (2 * p) : (16 + 2 * (p - 4))) + khalf * 8;
      const float2 fa = *(const float2*)(arow + k);
      const float2 fb = *(const float2*)(brow + k);
      afrag[2 * p]     = (_Float16)fa.x;
      afrag[2 * p + 1] = (_Float16)fa.y;
      bfrag[2 * p]     = (_Float16)(fb.x * bscale);
      bfrag[2 * p + 1] = (_Float16)(fb.y * bscale);
    }
    acc = __builtin_amdgcn_wmma_f32_16x16x32_f16(
        /*neg_a=*/false, afrag, /*neg_b=*/false, bfrag,
        /*c_mod=*/(short)0, acc, /*reuse_a=*/false, /*reuse_b=*/false);
  }

  // Epilogue: C/D layout -> lane l, vgpr r = element (m_base + r + 8*(l>>4), l&15)
  const int   col  = col_tile * 16 + r16;
  const float badd = (is_mod && col < VOUT) ? bias[(col < VOUT) ? col : 0] : 0.0f;
  #pragma unroll
  for (int r = 0; r < 8; ++r) {
    const int m = m_base + r + 8 * khalf;
    const int x = m >> 7;
    const int b = m & 127;
    out[(size_t)x * (VPAD * BB) + (size_t)col * BB + b] = acc[r] + badd;
  }
}

// ------------------- Phase 2: fused logsumexp + outputs --------------------
// grid (64,64) = (x,y); 128 threads = b. All ws reads coalesced across b.
__global__ __launch_bounds__(128) void edit_fuse(
    const float* __restrict__ ws,
    const int* __restrict__ target_ids,   // (63,128)
    const int* __restrict__ src_len,      // (128,)
    const int* __restrict__ tgt_len,      // (128,)
    float* __restrict__ out)
{
  const int b = threadIdx.x;
  const int x = blockIdx.x;
  const int y = blockIdx.y;

  const float* __restrict__ As = ws + 0 * (size_t)MAT_STRIDE + (size_t)x * (VPAD * BB);
  const float* __restrict__ Ai = ws + 1 * (size_t)MAT_STRIDE + (size_t)x * (VPAD * BB);
  const float* __restrict__ Cs = ws + 2 * (size_t)MAT_STRIDE + (size_t)y * (VPAD * BB);
  const float* __restrict__ Ci = ws + 3 * (size_t)MAT_STRIDE + (size_t)y * (VPAD * BB);

  const bool mask = (x < src_len[b]) && (y < tgt_len[b]);
  const int  t    = (y < LY - 1) ? target_ids[y * BB + b] : 0;

  float ms = -INFINITY, ss = 0.0f;   // sub head online logsumexp
  float mi = -INFINITY, si = 0.0f;   // ins head online logsumexp
  float s_t = 0.0f, i_t = 0.0f;      // logits at target id
  float s52 = 0.0f, i52 = 0.0f;      // logits at v = VOUT-1

  for (int v = 0; v < VOUT; ++v) {
    const float sv = As[v * BB + b] + Cs[v * BB + b];
    const float iv = Ai[v * BB + b] + Ci[v * BB + b];

    const float nms = fmaxf(ms, sv);
    ss = ss * __expf(ms - nms) + __expf(sv - nms);
    ms = nms;
    const float nmi = fmaxf(mi, iv);
    si = si * __expf(mi - nmi) + __expf(iv - nmi);
    mi = nmi;

    if (v == t)        { s_t = sv; i_t = iv; }
    if (v == VOUT - 1) { s52 = sv; i52 = iv; }
  }

  const float lses = ms + __logf(ss);
  const float lsei = mi + __logf(si);

  // outputs concatenated flat: del(64*64*128) | end(64*64*128) |
  //                            sub(64*63*128) | ins(64*63*128)
  const int xy = (x * LY + y) * BB + b;
  out[xy]                      = mask ? (s52 - lses) : 0.0f;
  out[LX * LY * BB + xy]       = mask ? (i52 - lsei) : 0.0f;
  if (y < LY - 1) {
    const int xy1 = (x * (LY - 1) + y) * BB + b;
    out[2 * LX * LY * BB + xy1]                       = mask ? (s_t - lses) : 0.0f;
    out[2 * LX * LY * BB + LX * (LY - 1) * BB + xy1]  = mask ? (i_t - lsei) : 0.0f;
  }
}

// ------------------------------- launcher ----------------------------------
extern "C" void kernel_launch(void* const* d_in, const int* in_sizes, int n_in,
                              void* d_out, int out_size, void* d_ws, size_t ws_size,
                              hipStream_t stream) {
  const float* prior  = (const float*)d_in[0];
  const float* modern = (const float*)d_in[1];
  const float* Wsub   = (const float*)d_in[2];
  const float* bsub   = (const float*)d_in[3];
  const float* Wins   = (const float*)d_in[4];
  const float* bins   = (const float*)d_in[5];
  const int*   tids   = (const int*)d_in[6];
  const int*   slen   = (const int*)d_in[7];
  const int*   tlen   = (const int*)d_in[8];
  float* out = (float*)d_out;
  float* wsf = (float*)d_ws;   // needs 4 * 524288 * 4B = 8 MB

  // Phase 1: 8192 wave-tiles (512 row-tiles x 4 col-tiles x 4 matrices)
  edit_gemm_wmma<<<dim3(1024), dim3(256), 0, stream>>>(
      prior, modern, Wsub, bsub, Wins, bins, wsf);

  // Phase 2: one block per (x,y), one lane per b
  edit_fuse<<<dim3(LX, LY), dim3(BB), 0, stream>>>(wsf, tids, slen, tlen, out);
}
